// SparseVoxelNetLiDAREncoder_73813307949091
// MI455X (gfx1250) — matrix operand order repaired and checked by reading, not compiled, not run on hardware
//
#include <hip/hip_runtime.h>

// SparseVoxelNet LiDAR encoder for gfx1250 (MI455X).
// All GEMM-like work (VFE2, 3D convs as implicit GEMM, BEV convs) runs on
// v_wmma_f32_16x16x32_bf16. Activations are channel-last bf16 so every
// 32-channel K-chunk is two contiguous 16B loads per lane, in contiguous k =
// order 16*half + s, the same for A and B. BN/ReLU/bias/mask fused in epilogues.
// Conv waves process NT=4 output-channel tiles per A load (4 WMMAs per A fetch)
// to amortize activation traffic and address arithmetic.
// Workspace requirement: ~130 MB (fits MI455X's 192MB L2 entirely).

typedef __bf16 bf16;
typedef __attribute__((ext_vector_type(16))) __bf16 v16bf;
typedef __attribute__((ext_vector_type(8)))  __bf16 v8bf;
typedef __attribute__((ext_vector_type(8)))  float  v8f;

#define GXX 100
#define GYY 100
#define GZZ 20
#define NDENSE 200000     // GXX*GYY*GZZ
#define MAXV 60000
#define MAXP 10

static __device__ __forceinline__ bf16 f2bf(float f){
  unsigned u = __builtin_bit_cast(unsigned, f);
  unsigned r = (u + 0x7FFFu + ((u >> 16) & 1u)) >> 16;   // round-to-nearest-even
  unsigned short us = (unsigned short)r;
  return __builtin_bit_cast(bf16, us);
}

static __device__ __forceinline__ v8f wmma_bf16(v16bf a, v16bf b, v8f c){
  // (neg_a, A, neg_b, B, c_mod, C, reuse_a, reuse_b)
  return __builtin_amdgcn_wmma_f32_16x16x32_bf16(false, a, false, b, (short)0, c, false, false);
}

static __device__ __forceinline__ v16bf cat16(v8bf lo, v8bf hi){
  return __builtin_shufflevector(lo, hi, 0,1,2,3,4,5,6,7,8,9,10,11,12,13,14,15);
}

// ---------------- utility kernels ----------------

__global__ void k_zero(unsigned* __restrict__ p, long long n){
  long long i = (long long)blockIdx.x * 256 + threadIdx.x;
  if (i < n) p[i] = 0u;
}

__global__ void k_fill_i32(int* __restrict__ p, long long n, int v){
  long long i = (long long)blockIdx.x * 256 + threadIdx.x;
  if (i < n) p[i] = v;
}

// transpose conv weights [N][Cin][KK] (f32) -> [N][KK*Cin] (bf16), k-major
__global__ void k_wconv(const float* __restrict__ src, bf16* __restrict__ dst,
                        int N, int Cin, int KK){
  long long i = (long long)blockIdx.x * 256 + threadIdx.x;
  long long tot = (long long)N * Cin * KK;
  if (i >= tot) return;
  int inner = Cin * KK;
  int n = (int)(i / inner);
  int r = (int)(i % inner);
  int kidx = r / Cin;
  int cin  = r % Cin;
  dst[i] = f2bf(src[((long long)n * Cin + cin) * KK + kidx]);
}

// ---------------- voxelization ----------------

__global__ void k_mark(const float* __restrict__ pts, int N, unsigned* __restrict__ occ){
  int i = blockIdx.x * 256 + threadIdx.x;
  if (i >= N) return;
  float px = pts[i*5+0], py = pts[i*5+1], pz = pts[i*5+2];
  if (px == 0.f) return;
  int ix = (int)floorf((px + 51.2f) / 1.024f);
  int iy = (int)floorf((py + 51.2f) / 1.024f);
  int iz = (int)floorf((pz + 5.0f)  / 0.4f);
  if (ix < 0 || ix >= GXX || iy < 0 || iy >= GYY || iz < 0 || iz >= GZZ) return;
  occ[ix * (GYY*GZZ) + iy * GZZ + iz] = 1u;
}

__global__ __launch_bounds__(256) void k_blocksum(const unsigned* __restrict__ occ,
                                                  unsigned* __restrict__ bs, int n){
  __shared__ unsigned wsum[8];
  int i = blockIdx.x * 256 + threadIdx.x;
  unsigned f = (i < n) ? occ[i] : 0u;
  unsigned long long bal = __ballot(f != 0u);
  int lane = threadIdx.x & 31, wid = threadIdx.x >> 5;
  if (lane == 0) wsum[wid] = (unsigned)__popcll(bal);
  __syncthreads();
  if (threadIdx.x == 0){
    unsigned s = 0;
    for (int w = 0; w < 8; w++) s += wsum[w];
    bs[blockIdx.x] = s;
  }
}

__global__ void k_topscan(const unsigned* __restrict__ bs, unsigned* __restrict__ bo, int nb){
  if (blockIdx.x == 0 && threadIdx.x == 0){
    unsigned run = 0;
    for (int b = 0; b < nb; b++){ bo[b] = run; run += bs[b]; }
  }
}

__global__ __launch_bounds__(256) void k_assign(const unsigned* __restrict__ occ,
                                                const unsigned* __restrict__ bo, int n,
                                                int* __restrict__ map, int* __restrict__ coords,
                                                float* __restrict__ vmask){
  __shared__ unsigned wsum[8];
  int i = blockIdx.x * 256 + threadIdx.x;
  unsigned f = (i < n) ? occ[i] : 0u;
  unsigned long long bal = __ballot(f != 0u);
  int lane = threadIdx.x & 31, wid = threadIdx.x >> 5;
  if (lane == 0) wsum[wid] = (unsigned)__popcll(bal);
  __syncthreads();
  unsigned woff = 0;
  for (int w = 0; w < wid; w++) woff += wsum[w];
  unsigned pre = (unsigned)__popcll(bal & ((1ull << lane) - 1ull));
  if (f){
    unsigned slot = bo[blockIdx.x] + woff + pre;   // ascending-hash order == sorted order
    if (slot < MAXV){
      map[i] = (int)slot;
      int ix = i / (GYY*GZZ); int r = i % (GYY*GZZ); int iy = r / GZZ; int iz = r % GZZ;
      coords[slot*3+0] = ix; coords[slot*3+1] = iy; coords[slot*3+2] = iz;
      vmask[slot] = 1.f;
    }
  }
}

__global__ void k_scatter(const float* __restrict__ pts, int N, const int* __restrict__ map,
                          unsigned* __restrict__ cnt, float* __restrict__ vox){
  int i = blockIdx.x * 256 + threadIdx.x;
  if (i >= N) return;
  float px = pts[i*5+0], py = pts[i*5+1], pz = pts[i*5+2];
  if (px == 0.f) return;
  int ix = (int)floorf((px + 51.2f) / 1.024f);
  int iy = (int)floorf((py + 51.2f) / 1.024f);
  int iz = (int)floorf((pz + 5.0f)  / 0.4f);
  if (ix < 0 || ix >= GXX || iy < 0 || iy >= GYY || iz < 0 || iz >= GZZ) return;
  int slot = map[ix * (GYY*GZZ) + iy * GZZ + iz];
  if (slot < 0) return;
  unsigned rank = atomicAdd(&cnt[slot], 1u);
  if (rank < MAXP){
    float* d = vox + ((size_t)slot * MAXP + rank) * 5;
    for (int k = 0; k < 5; k++) d[k] = pts[i*5+k];
  }
}

// ---------------- VFE (one wave per voxel; VFE2 = 8 bf16 WMMAs) ----------------

__global__ __launch_bounds__(32) void k_vfe(
    const float* __restrict__ vox, const int* __restrict__ coords,
    const float* __restrict__ vmask,
    const float* __restrict__ w1, const float* __restrict__ b1,
    const float* __restrict__ g1, const float* __restrict__ be1,
    const bf16*  __restrict__ w2t, const float* __restrict__ b2,
    const float* __restrict__ g2, const float* __restrict__ be2,
    bf16* __restrict__ dense0, float* __restrict__ m0)
{
  int v = blockIdx.x;
  if (vmask[v] == 0.f) return;          // wave-uniform
  int L = threadIdx.x;
  __shared__ __align__(16) bf16 lp[16 * 32];

  // VFE1: lane L computes channel L for all 10 (possibly zero-padded) points
  float w[5];
  #pragma unroll
  for (int k = 0; k < 5; k++) w[k] = w1[L*5 + k];
  float bb = b1[L], gg = g1[L], bt = be1[L];
  const float* pv = vox + (size_t)v * MAXP * 5;
  for (int j = 0; j < MAXP; j++){
    float s = bb;
    #pragma unroll
    for (int k = 0; k < 5; k++) s += pv[j*5 + k] * w[k];
    lp[j*32 + L] = f2bf(fmaxf(s*gg + bt, 0.f));
  }
  float dflt = fmaxf(bb*gg + bt, 0.f);  // empty-point row value
  for (int j = MAXP; j < 16; j++) lp[j*32 + L] = f2bf(dflt);
  __syncthreads();

  // pack WMMA A (16x32 bf16, k = 16*half + t, same order as B)
  int lm = L & 15, half = L >> 4, kb8 = half * 8;
  v16bf A;
  #pragma unroll
  for (int t = 0; t < 8; t++){
    A[t]     = lp[lm*32 + half*16 + t];
    A[8 + t] = lp[lm*32 + half*16 + 8 + t];
  }

  int ix = coords[v*3+0], iy = coords[v*3+1], iz = coords[v*3+2];
  int s = (iz * GYY + iy) * GXX + ix;

  for (int nt = 0; nt < 8; nt++){
    int n = nt * 16 + lm;
    v8bf blo = *(const v8bf*)(w2t + n*32 + half*16);
    v8bf bhi = *(const v8bf*)(w2t + n*32 + half*16 + 8);
    v16bf B = cat16(blo, bhi);
    v8f acc;
    #pragma unroll
    for (int t = 0; t < 8; t++) acc[t] = 0.f;
    acc = wmma_bf16(A, B, acc);
    // epilogue: BN + ReLU, then max over the 16 point-rows
    float cb = b2[n], cg = g2[n], ct = be2[n];
    float mx = -3.4e38f;
    #pragma unroll
    for (int t = 0; t < 8; t++) mx = fmaxf(mx, fmaxf((acc[t] + cb)*cg + ct, 0.f));
    float o = fmaxf(mx, __shfl_xor(mx, 16, 32));
    if (half == 0) dense0[(size_t)s * 128 + n] = f2bf(o);
  }
  if (L == 0) m0[s] = 1.f;
}

// ---------------- occupancy pool (SparseConv mask, 3x3x3 s2 p1) ----------------

__global__ void k_pool(const float* __restrict__ src, float* __restrict__ dst,
                       int Di, int Hi, int Wi, int Do, int Ho, int Wo){
  int i = blockIdx.x * 256 + threadIdx.x;
  int M = Do * Ho * Wo;
  if (i >= M) return;
  int z = i / (Ho*Wo); int r = i % (Ho*Wo); int y = r / Wo; int x = r % Wo;
  float m = 0.f;
  for (int dz = -1; dz <= 1; dz++){
    int iz = 2*z + dz; if (iz < 0 || iz >= Di) continue;
    for (int dy = -1; dy <= 1; dy++){
      int iy = 2*y + dy; if (iy < 0 || iy >= Hi) continue;
      for (int dx = -1; dx <= 1; dx++){
        int ix = 2*x + dx; if (ix < 0 || ix >= Wi) continue;
        m = fmaxf(m, src[(iz*Hi + iy)*Wi + ix]);
      }
    }
  }
  dst[i] = m;
}

// ---------------- implicit-GEMM 3D conv (3x3x3, pad1), bf16 WMMA ----------------
// one wave per 16-site x (NT*16)-outC tile: A held in registers, NT WMMAs per A
// fetch; mask-folded epilogue; all-masked tiles skipped (wave-uniform ballot)

template<int NT>
__global__ __launch_bounds__(32) void k_conv3d(
    const bf16* __restrict__ in, const bf16* __restrict__ wt,
    const float* __restrict__ g, const float* __restrict__ be,
    const float* __restrict__ mask, bf16* __restrict__ out,
    int Din, int Hin, int Win, int Dout, int Hout, int Wout,
    int Cin, int Cout, int stride)
{
  const int L = threadIdx.x, lm = L & 15, half = L >> 4, kb8 = half * 8;
  const int HW = Hout * Wout, M = Dout * HW;
  int site = blockIdx.x * 16 + lm;
  int sc = site < M ? site : M - 1;
  int z = sc / HW; int r = sc % HW; int y = r / Wout; int x = r % Wout;
  const int n0 = blockIdx.y * (NT * 16) + lm;

  // sparsity: skip tile if all 16 output sites are masked out (wave-uniform)
  bool act = (site < M) && (mask[sc] != 0.f);
  unsigned long long bal = __ballot(act);
  if (bal == 0ull){
    #pragma unroll
    for (int nt = 0; nt < NT; nt++)
      #pragma unroll
      for (int t = 0; t < 8; t++){
        int sm = blockIdx.x*16 + t + half*8;
        if (sm < M) out[(size_t)sm * Cout + n0 + nt*16] = f2bf(0.f);
      }
    return;
  }

  const bf16* wrow[NT];
  #pragma unroll
  for (int nt = 0; nt < NT; nt++) wrow[nt] = wt + (size_t)(n0 + nt*16) * (27 * Cin);

  v8f acc[NT];
  #pragma unroll
  for (int nt = 0; nt < NT; nt++)
    #pragma unroll
    for (int t = 0; t < 8; t++) acc[nt][t] = 0.f;

  for (int kz = 0; kz < 3; kz++){
    int iz = z * stride + kz - 1;
    for (int ky = 0; ky < 3; ky++){
      int iy = y * stride + ky - 1;
      for (int kx = 0; kx < 3; kx++){
        int ix = x * stride + kx - 1;
        bool inb = (iz >= 0) & (iz < Din) & (iy >= 0) & (iy < Hin) & (ix >= 0) & (ix < Win);
        const bf16* arow = in + (size_t)((iz * Hin + iy) * Win + ix) * Cin;
        const int koff = ((kz*3 + ky)*3 + kx) * Cin;
        for (int c0 = 0; c0 < Cin; c0 += 32){
          v16bf A;
          if (inb){
            v8bf lo = *(const v8bf*)(arow + c0 + half*16);
            v8bf hi = *(const v8bf*)(arow + c0 + half*16 + 8);
            A = cat16(lo, hi);
          } else {
            #pragma unroll
            for (int t = 0; t < 16; t++) A[t] = f2bf(0.f);
          }
          #pragma unroll
          for (int nt = 0; nt < NT; nt++){
            const bf16* brow = wrow[nt] + koff + c0 + half*16;
            v8bf b0 = *(const v8bf*)(brow);
            v8bf b1 = *(const v8bf*)(brow + 8);
            acc[nt] = wmma_bf16(A, cat16(b0, b1), acc[nt]);
          }
        }
      }
    }
  }
  #pragma unroll
  for (int nt = 0; nt < NT; nt++){
    int n = n0 + nt*16;
    float gg = g[n], bb = be[n];
    #pragma unroll
    for (int t = 0; t < 8; t++){
      int sm = blockIdx.x*16 + t + half*8;
      if (sm < M){
        float v = fmaxf(acc[nt][t]*gg + bb, 0.f) * mask[sm];
        out[(size_t)sm * Cout + n] = f2bf(v);
      }
    }
  }
}

// ---------------- implicit-GEMM 2D BEV conv (3x3, pad1, s1), bf16 WMMA ----------
// virtual channels cv map to plane = cv/Cblock of a [planes][H*W][Cblock] input
// (handles the (D,C)->D*C reshape with zero data movement); NT n-tiles per wave

template<int NT>
__global__ __launch_bounds__(32) void k_conv2d(
    const bf16* __restrict__ in, const bf16* __restrict__ wt,
    const float* __restrict__ g, const float* __restrict__ be,
    const float* __restrict__ bias,
    bf16* __restrict__ outb, float* __restrict__ outf,
    int H, int W, int Cv, int Cblock, int Cout)
{
  const int L = threadIdx.x, lm = L & 15, half = L >> 4, kb8 = half * 8;
  const int M = H * W;
  int site = blockIdx.x * 16 + lm;
  int sc = site < M ? site : M - 1;
  int y = sc / W, x = sc % W;
  const int n0 = blockIdx.y * (NT * 16) + lm;

  const bf16* wrow[NT];
  #pragma unroll
  for (int nt = 0; nt < NT; nt++) wrow[nt] = wt + (size_t)(n0 + nt*16) * 9 * Cv;

  v8f acc[NT];
  #pragma unroll
  for (int nt = 0; nt < NT; nt++)
    #pragma unroll
    for (int t = 0; t < 8; t++) acc[nt][t] = 0.f;

  for (int ky = 0; ky < 3; ky++){
    int iy = y + ky - 1;
    for (int kx = 0; kx < 3; kx++){
      int ix = x + kx - 1;
      bool inb = (iy >= 0) & (iy < H) & (ix >= 0) & (ix < W);
      const int koff = (ky*3 + kx) * Cv;
      for (int c0 = 0; c0 < Cv; c0 += 32){
        v16bf A;
        if (inb){
          int plane = c0 / Cblock;
          int offc  = c0 - plane * Cblock;
          const bf16* arow = in + ((size_t)plane * M + iy * W + ix) * Cblock + offc;
          v8bf lo = *(const v8bf*)(arow + half*16);
          v8bf hi = *(const v8bf*)(arow + half*16 + 8);
          A = cat16(lo, hi);
        } else {
          #pragma unroll
          for (int t = 0; t < 16; t++) A[t] = f2bf(0.f);
        }
        #pragma unroll
        for (int nt = 0; nt < NT; nt++){
          const bf16* brow = wrow[nt] + koff + c0 + half*16;
          v8bf b0 = *(const v8bf*)(brow);
          v8bf b1 = *(const v8bf*)(brow + 8);
          acc[nt] = wmma_bf16(A, cat16(b0, b1), acc[nt]);
        }
      }
    }
  }
  #pragma unroll
  for (int nt = 0; nt < NT; nt++){
    int n = n0 + nt*16;
    float gg = g[n], bb = be[n], bi = bias[n];
    #pragma unroll
    for (int t = 0; t < 8; t++){
      int sm = blockIdx.x*16 + t + half*8;
      if (sm < M){
        float v = fmaxf((acc[nt][t] + bi)*gg + bb, 0.f);
        if (outf) outf[(size_t)n * M + sm] = v;                 // NCHW f32 final output
        else      outb[(size_t)sm * Cout + n] = f2bf(v);        // channel-last bf16
      }
    }
  }
}

// ---------------- host ----------------

extern "C" void kernel_launch(void* const* d_in, const int* in_sizes, int n_in,
                              void* d_out, int out_size, void* d_ws, size_t ws_size,
                              hipStream_t stream) {
  const float* pts      = (const float*)d_in[0];
  const int    Npts     = in_sizes[0] / 5;
  const float* vfe1_w   = (const float*)d_in[1];
  const float* vfe1_b   = (const float*)d_in[2];
  const float* vfe1_g   = (const float*)d_in[3];
  const float* vfe1_be  = (const float*)d_in[4];
  const float* vfe2_w   = (const float*)d_in[5];
  const float* vfe2_b   = (const float*)d_in[6];
  const float* vfe2_g   = (const float*)d_in[7];
  const float* vfe2_be  = (const float*)d_in[8];
  const float* sp1_w    = (const float*)d_in[9];
  const float* sp1_g    = (const float*)d_in[10];
  const float* sp1_be   = (const float*)d_in[11];
  const float* sp2_w    = (const float*)d_in[12];
  const float* sp2_g    = (const float*)d_in[13];
  const float* sp2_be   = (const float*)d_in[14];
  const float* sp3_w    = (const float*)d_in[15];
  const float* sp3_g    = (const float*)d_in[16];
  const float* sp3_be   = (const float*)d_in[17];
  const float* sp4_w    = (const float*)d_in[18];
  const float* sp4_g    = (const float*)d_in[19];
  const float* sp4_be   = (const float*)d_in[20];
  const float* bev1_w   = (const float*)d_in[21];
  const float* bev1_b   = (const float*)d_in[22];
  const float* bev1_g   = (const float*)d_in[23];
  const float* bev1_be  = (const float*)d_in[24];
  const float* bev2_w   = (const float*)d_in[25];
  const float* bev2_b   = (const float*)d_in[26];
  const float* bev2_g   = (const float*)d_in[27];
  const float* bev2_be  = (const float*)d_in[28];

  char* base = (char*)d_ws;
  size_t off = 0;
  auto alloc = [&](size_t bytes) -> void* {
    void* p = base + off;
    off = (off + bytes + 255) & ~(size_t)255;
    return p;
  };

  unsigned* occ    = (unsigned*)alloc((size_t)NDENSE * 4);
  unsigned* bs     = (unsigned*)alloc(1024 * 4);
  unsigned* bo     = (unsigned*)alloc(1024 * 4);
  int*      map    = (int*)     alloc((size_t)NDENSE * 4);
  unsigned* cnt    = (unsigned*)alloc((size_t)MAXV * 4);
  int*      coords = (int*)     alloc((size_t)MAXV * 3 * 4);
  float*    vmask  = (float*)   alloc((size_t)MAXV * 4);
  float*    vox    = (float*)   alloc((size_t)MAXV * MAXP * 5 * 4);
  float*    m0     = (float*)   alloc((size_t)NDENSE * 4);
  float*    m1     = (float*)   alloc((size_t)25000 * 4);
  float*    m2     = (float*)   alloc((size_t)3125 * 4);
  bf16*     dense0 = (bf16*)    alloc((size_t)NDENSE * 128 * 2);
  bf16*     o1     = (bf16*)    alloc((size_t)NDENSE * 64 * 2);
  bf16*     o2     = (bf16*)    alloc((size_t)25000 * 128 * 2);
  bf16*     o3     = (bf16*)    alloc((size_t)25000 * 128 * 2);
  bf16*     o4     = (bf16*)    alloc((size_t)3125 * 256 * 2);
  bf16*     bv1    = (bf16*)    alloc((size_t)625 * 512 * 2);
  bf16*     w2t    = (bf16*)    alloc((size_t)128 * 32 * 2);
  bf16*     sp1t   = (bf16*)    alloc((size_t)64 * 3456 * 2);
  bf16*     sp2t   = (bf16*)    alloc((size_t)128 * 1728 * 2);
  bf16*     sp3t   = (bf16*)    alloc((size_t)128 * 3456 * 2);
  bf16*     sp4t   = (bf16*)    alloc((size_t)256 * 3456 * 2);
  bf16*     bv1t   = (bf16*)    alloc((size_t)512 * 11520 * 2);
  bf16*     bv2t   = (bf16*)    alloc((size_t)256 * 4608 * 2);
  (void)n_in; (void)out_size; (void)ws_size;

  auto Z = [&](void* p, size_t words){
    k_zero<<<dim3((unsigned)((words + 255) / 256)), dim3(256), 0, stream>>>((unsigned*)p, (long long)words);
  };
  Z(occ,   NDENSE);
  Z(cnt,   MAXV);
  Z(vmask, MAXV);
  Z(vox,   (size_t)MAXV * MAXP * 5);
  Z(m0,    NDENSE);
  Z(dense0,(size_t)NDENSE * 64);   // NDENSE*128 bf16 == NDENSE*64 dwords
  k_fill_i32<<<dim3((NDENSE + 255) / 256), dim3(256), 0, stream>>>(map, NDENSE, -1);

  auto WC = [&](const float* s, bf16* d, int N, int Ci, int KK){
    long long tot = (long long)N * Ci * KK;
    k_wconv<<<dim3((unsigned)((tot + 255) / 256)), dim3(256), 0, stream>>>(s, d, N, Ci, KK);
  };
  WC(vfe2_w, w2t, 128, 32, 1);
  WC(sp1_w, sp1t, 64, 128, 27);
  WC(sp2_w, sp2t, 128, 64, 27);
  WC(sp3_w, sp3t, 128, 128, 27);
  WC(sp4_w, sp4t, 256, 128, 27);
  WC(bev1_w, bv1t, 512, 1280, 9);
  WC(bev2_w, bv2t, 256, 512, 9);

  // voxelize: mark -> block sums -> top scan -> slot assign -> scatter points
  const int NB = (NDENSE + 255) / 256;   // 782
  k_mark<<<dim3((Npts + 255) / 256), dim3(256), 0, stream>>>(pts, Npts, occ);
  k_blocksum<<<dim3(NB), dim3(256), 0, stream>>>(occ, bs, NDENSE);
  k_topscan<<<dim3(1), dim3(1), 0, stream>>>(bs, bo, NB);
  k_assign<<<dim3(NB), dim3(256), 0, stream>>>(occ, bo, NDENSE, map, coords, vmask);
  k_scatter<<<dim3((Npts + 255) / 256), dim3(256), 0, stream>>>(pts, Npts, map, cnt, vox);

  // VFE (WMMA) + scatter to dense channel-last bf16 grid
  k_vfe<<<dim3(MAXV), dim3(32), 0, stream>>>(vox, coords, vmask,
      vfe1_w, vfe1_b, vfe1_g, vfe1_be, w2t, vfe2_b, vfe2_g, vfe2_be, dense0, m0);

  // sp1: SubMConv3d 128->64, s1  (20,100,100); 64 outC = one NT=4 group
  k_conv3d<4><<<dim3(12500, 1), dim3(32), 0, stream>>>(dense0, sp1t, sp1_g, sp1_be, m0, o1,
      GZZ, GYY, GXX, GZZ, GYY, GXX, 128, 64, 1);
  // m1 = pool(m0): (20,100,100)->(10,50,50)
  k_pool<<<dim3((25000 + 255) / 256), dim3(256), 0, stream>>>(m0, m1, GZZ, GYY, GXX, 10, 50, 50);
  // sp2: SparseConv3d 64->128, s2 -> (10,50,50); 128 outC = 2 NT=4 groups
  k_conv3d<4><<<dim3((25000 + 15) / 16, 2), dim3(32), 0, stream>>>(o1, sp2t, sp2_g, sp2_be, m1, o2,
      GZZ, GYY, GXX, 10, 50, 50, 64, 128, 2);
  // sp3: SubMConv3d 128->128, s1 (10,50,50)
  k_conv3d<4><<<dim3((25000 + 15) / 16, 2), dim3(32), 0, stream>>>(o2, sp3t, sp3_g, sp3_be, m1, o3,
      10, 50, 50, 10, 50, 50, 128, 128, 1);
  // m2 = pool(m1): (10,50,50)->(5,25,25)
  k_pool<<<dim3((3125 + 255) / 256), dim3(256), 0, stream>>>(m1, m2, 10, 50, 50, 5, 25, 25);
  // sp4: SparseConv3d 128->256, s2 -> (5,25,25); 256 outC = 4 NT=4 groups
  k_conv3d<4><<<dim3((3125 + 15) / 16, 4), dim3(32), 0, stream>>>(o3, sp4t, sp4_g, sp4_be, m2, o4,
      10, 50, 50, 5, 25, 25, 128, 256, 2);

  // bev1: 1280->512, 3x3 p1 over 25x25 (virtual channels over 5 depth planes of 256)
  k_conv2d<4><<<dim3((625 + 15) / 16, 8), dim3(32), 0, stream>>>(o4, bv1t, bev1_g, bev1_be, bev1_b,
      bv1, (float*)nullptr, 25, 25, 1280, 256, 512);
  // bev2: 512->256, 3x3 p1; write fp32 NCHW straight to d_out
  k_conv2d<4><<<dim3((625 + 15) / 16, 4), dim3(32), 0, stream>>>(bv1, bv2t, bev2_g, bev2_be, bev2_b,
      (bf16*)nullptr, (float*)d_out, 25, 25, 512, 512, 256);
}
